// BiLSTM_CRF_80547816669585
// MI455X (gfx1250) — compile-verified, matrix operand order checked by reading, not run on hardware
//
#include <hip/hip_runtime.h>

typedef _Float16 v16h  __attribute__((ext_vector_type(16)));
typedef _Float16 f16x8 __attribute__((ext_vector_type(8)));
typedef float    v8f   __attribute__((ext_vector_type(8)));
typedef int      v4i   __attribute__((vector_size(16)));

#define B_    128
#define T_    512
#define K_    20
#define DPAD  512
#define H_    256
#define G_    1024          // 4H
#define NROW  (T_*B_)       // 65536

// workspace byte offsets
#define OFF_EMB  ((size_t)0)            // [65536][512] f16
#define OFF_WIH  ((size_t)67108864)     // packed w_ih (both dirs) f16
#define OFF_WHH  ((size_t)69206016)     // packed w_hh (both dirs) f16
#define OFF_BIAS ((size_t)70254592)     // [2][1024] f32 (b_ih+b_hh)
#define OFF_PRE  ((size_t)70262784)     // [2][T][B][G] f32
#define OFF_H    ((size_t)607133696)    // [2][T][B][H] f32
#define OFF_EM   ((size_t)741351424)    // [T][B][32] f32 (K padded)
#define OFF_NLL  ((size_t)749740032)    // [128] f32 per-batch nll

// ---- gfx1250 async global->LDS path (ASYNCcnt), guarded so a missing builtin
// falls back to the known-good synchronous staging copy.
#if defined(__has_builtin)
#  if __has_builtin(__builtin_amdgcn_global_load_async_to_lds_b128)
#    define HAVE_ASYNC_LDS 1
#  endif
#  if __has_builtin(__builtin_amdgcn_s_wait_asynccnt)
#    define HAVE_WAIT_ASYNC 1
#  endif
#endif
#ifndef HAVE_ASYNC_LDS
#  define HAVE_ASYNC_LDS 0
#endif
#ifndef HAVE_WAIT_ASYNC
#  define HAVE_WAIT_ASYNC 0
#endif

// builtin signature: (v4i addrspace(1)* src, v4i addrspace(3)* dst, imm off, imm cpol)
typedef __attribute__((address_space(1))) v4i* gasp_t;
typedef __attribute__((address_space(3))) v4i* lasp_t;

__device__ __forceinline__ void wait_async_zero() {
#if HAVE_ASYNC_LDS
#  if HAVE_WAIT_ASYNC
  __builtin_amdgcn_s_wait_asynccnt(0);
#  else
  asm volatile("s_wait_asynccnt 0x0" ::: "memory");
#  endif
#endif
}

union AF { v16h v; f16x8 h[2]; };

// 16-bit A-matrix 16x32 lane layout (ISA 7.12.2): lanes 0-15 hold row M=lane,
// K chunks {0..7,16..23}; lanes 16-31 hold row M=lane-16, K chunks {8..15,24..31}.
__device__ __forceinline__ int kmap(int lane, int j) {
  int koff = (lane < 16) ? 0 : 8;
  return (j < 8) ? (koff + j) : (16 + koff + (j - 8));
}

// ---------------- embedding gather: concat(char,pinyin,tag) -> f16, D padded to 512
__global__ void embed_gather(const int* __restrict__ x, const int* __restrict__ ptg,
                             const int* __restrict__ pyt,
                             const float* __restrict__ ce, const float* __restrict__ te,
                             const float* __restrict__ pe, _Float16* __restrict__ emb) {
  int idx = blockIdx.x * 256 + threadIdx.x;      // < NROW*DPAD
  int d = idx & (DPAD - 1);
  int r = idx >> 9;                              // r = t*B + b
  int t = r >> 7, b = r & 127;
  float v = 0.f;
  if (d < 300)      v = ce[(size_t)x[b * T_ + t] * 300 + d];
  else if (d < 400) v = pe[(size_t)pyt[b * T_ + t] * 100 + (d - 300)];
  else if (d < 450) v = te[(size_t)ptg[b * T_ + t] * 50 + (d - 400)];
  emb[(size_t)r * DPAD + d] = (_Float16)v;
}

// ---------------- pack W_ih (both dirs) into B-fragment layout, K padded 450->512
__global__ void pack_wih(const float* __restrict__ wf, const float* __restrict__ wb,
                         _Float16* __restrict__ out) {
  int idx = blockIdx.x * 256 + threadIdx.x;      // < 128*16*32*16
  int j = idx & 15, lane = (idx >> 4) & 31, ki = (idx >> 9) & 15, ct = idx >> 13;
  int n = ct * 16 + (lane & 15);                 // output column (dir*1024+g)
  int dir = n >> 10, g = n & 1023;
  int k = ki * 32 + kmap(lane, j);
  const float* w = dir ? wb : wf;
  out[idx] = (_Float16)((k < 450) ? w[(size_t)g * 450 + k] : 0.f);
}

// ---------------- pack W_hh (both dirs) into B-fragment layout
__global__ void pack_whh(const float* __restrict__ wf, const float* __restrict__ wb,
                         _Float16* __restrict__ out) {
  int idx = blockIdx.x * 256 + threadIdx.x;      // < 2*64*8*32*16
  int j = idx & 15, lane = (idx >> 4) & 31, ki = (idx >> 9) & 7;
  int gt = (idx >> 12) & 63, dir = idx >> 18;
  int g = gt * 16 + (lane & 15);
  int k = ki * 32 + kmap(lane, j);
  const float* w = dir ? wb : wf;
  out[idx] = (_Float16)w[(size_t)g * H_ + k];
}

__global__ void bias_comb(const float* bihf, const float* bhhf,
                          const float* bihb, const float* bhhb, float* out) {
  int idx = blockIdx.x * 256 + threadIdx.x;      // < 2048
  int dir = idx >> 10, g = idx & 1023;
  out[idx] = dir ? (bihb[g] + bhhb[g]) : (bihf[g] + bhhf[g]);
}

// ---------------- input-projection GEMM, LDS-staged full-K tiles
// block = 8 waves; tile M=128, N=128, K=512; LDS: A 128KB + B 128KB (of 320KB WGP LDS)
// staging uses GLOBAL_LOAD_ASYNC_TO_LDS_B128 (ASYNCcnt) when available
// grid = (2048/128, 65536/128)
__global__ __launch_bounds__(256) void gemm_inproj(const _Float16* __restrict__ emb,
                                                   const _Float16* __restrict__ wihP,
                                                   const float* __restrict__ bias,
                                                   float* __restrict__ pre) {
  extern __shared__ __align__(16) _Float16 lds[];
  _Float16* As = lds;                       // [128][512] row-major f16
  _Float16* Bs = lds + 128 * 512;           // fragment-packed f16
  int tid = threadIdx.x, lane = tid & 31, w = tid >> 5;
  int m0 = blockIdx.y * 128;
  int colb = blockIdx.x;                    // 128-col block
  // cooperative global->LDS staging (unique data loaded exactly once per block)
  const f16x8* srcA = (const f16x8*)(emb + (size_t)m0 * DPAD);
  f16x8* dstA = (f16x8*)As;
  const f16x8* srcB = (const f16x8*)(wihP + (size_t)colb * (8 * 16 * 32 * 16));
  f16x8* dstB = (f16x8*)Bs;
#if HAVE_ASYNC_LDS
  for (int i = tid; i < 8192; i += 256)
    __builtin_amdgcn_global_load_async_to_lds_b128(
        (gasp_t)(srcA + i), (lasp_t)(dstA + i), 0, 0);
  for (int i = tid; i < 8192; i += 256)
    __builtin_amdgcn_global_load_async_to_lds_b128(
        (gasp_t)(srcB + i), (lasp_t)(dstB + i), 0, 0);
  wait_async_zero();
#else
  for (int i = tid; i < 8192; i += 256) dstA[i] = srcA[i];
  for (int i = tid; i < 8192; i += 256) dstB[i] = srcB[i];
#endif
  __syncthreads();

  int rw = w & 3, cw = w >> 2;              // 4 row groups x 2 col groups
  int koff = (lane < 16) ? 0 : 8, mlo = koff, nl = lane & 15;
  v8f acc[2][4];
#pragma unroll
  for (int ci = 0; ci < 4; ci++) {
    float bv = bias[colb * 128 + cw * 64 + ci * 16 + nl];
#pragma unroll
    for (int rt = 0; rt < 2; rt++)
#pragma unroll
      for (int e = 0; e < 8; e++) acc[rt][ci][e] = bv;
  }
#pragma unroll 2
  for (int ki = 0; ki < 16; ++ki) {
    AF a[2];
#pragma unroll
    for (int rt = 0; rt < 2; rt++) {
      const _Float16* ar = As + (rw * 32 + rt * 16 + nl) * DPAD + ki * 32;
      a[rt].h[0] = *(const f16x8*)(ar + koff);
      a[rt].h[1] = *(const f16x8*)(ar + 16 + koff);
    }
#pragma unroll
    for (int ci = 0; ci < 4; ci++) {
      int ctl = cw * 4 + ci;
      const v16h* bp = (const v16h*)(Bs + (((ctl * 16 + ki) * 32 + lane) << 4));
      v16h bfrag = *bp;
#pragma unroll
      for (int rt = 0; rt < 2; rt++)
        acc[rt][ci] = __builtin_amdgcn_wmma_f32_16x16x32_f16(
            false, a[rt].v, false, bfrag, (short)0, acc[rt][ci], false, false);
    }
  }
  int dir = (colb * 128) >> 10;             // uniform per block
  float* preD = pre + (size_t)dir * ((size_t)NROW * G_);
#pragma unroll
  for (int rt = 0; rt < 2; rt++)
#pragma unroll
    for (int ci = 0; ci < 4; ci++) {
      int g = (colb * 128 + cw * 64 + ci * 16 + nl) & 1023;
#pragma unroll
      for (int v = 0; v < 8; v++) {
        int r = m0 + rw * 32 + rt * 16 + v + mlo;
        preD[(size_t)r * G_ + g] = acc[rt][ci][v];
      }
    }
}

// ---------------- BiLSTM recurrence, batch-split: 16 independent workgroups
// (2 dirs x 8 batch-groups of 16). Wave w owns 32 h-cols; W_hh fragments are
// read with zero intra-WG redundancy (512KB/step, L2-hot). h double-buffered
// in 16KB static LDS; c-state in VGPRs; one barrier per step.
__global__ __launch_bounds__(256) void lstm_recurrent(const _Float16* __restrict__ whhP,
                                                      const float* __restrict__ pre,
                                                      float* __restrict__ hOut) {
  int dir = blockIdx.x >> 3, bg = blockIdx.x & 7;   // batch rows bg*16..+16
  __shared__ _Float16 hbuf[2][16 * H_];             // 2 x 8KB
  int tid = threadIdx.x, lane = tid & 31, w = tid >> 5;  // w: h-cols w*32..+32
  int koff = (lane < 16) ? 0 : 8, mlo = koff, nl = lane & 15;
  for (int i = tid; i < 2 * 16 * H_; i += 256) ((_Float16*)hbuf)[i] = (_Float16)0.f;
  __syncthreads();
  v8f cst[2];
#pragma unroll
  for (int s = 0; s < 2; s++)
#pragma unroll
    for (int e = 0; e < 8; e++) cst[s][e] = 0.f;
  const float*    preB = pre  + (size_t)dir * ((size_t)NROW * G_);
  float*          hD   = hOut + (size_t)dir * ((size_t)NROW * H_);
  const _Float16* whhD = whhP + (size_t)dir * (64 * 8 * 32 * 16);
  for (int step = 0; step < T_; ++step) {
    int tt = dir ? (T_ - 1 - step) : step;
    const _Float16* rB = hbuf[step & 1];
    _Float16*       wB = hbuf[(step & 1) ^ 1];
    const float* preT = preB + ((size_t)tt * B_ + bg * 16) * G_;
    // prefetch next step's pre-activations into cache (global_prefetch_b8)
    if (step + 1 < T_) {
      int tn = dir ? (T_ - 2 - step) : (step + 1);
      const char* pn = (const char*)(preB + ((size_t)tn * B_ + bg * 16) * G_);
      __builtin_prefetch(pn + tid * 256, 0, 1);
    }
    v8f ac[2][4];
#pragma unroll
    for (int s = 0; s < 2; s++)
#pragma unroll
      for (int gi = 0; gi < 4; gi++) {
        const float* pp = preT + gi * 256 + w * 32 + s * 16 + nl;
#pragma unroll
        for (int v = 0; v < 8; v++) ac[s][gi][v] = pp[(size_t)(v + mlo) * G_];
      }
    const _Float16* aRow = rB + nl * H_;
#pragma unroll 1
    for (int ki = 0; ki < 8; ++ki) {
      AF a;
      a.h[0] = *(const f16x8*)(aRow + ki * 32 + koff);
      a.h[1] = *(const f16x8*)(aRow + ki * 32 + 16 + koff);
#pragma unroll
      for (int s = 0; s < 2; s++)
#pragma unroll
        for (int gi = 0; gi < 4; gi++) {
          int gt = gi * 16 + w * 2 + s;
          const v16h* bp = (const v16h*)(whhD + (((size_t)gt * 8 + ki) * 32 + lane) * 16);
          ac[s][gi] = __builtin_amdgcn_wmma_f32_16x16x32_f16(
              false, a.v, false, *bp, (short)0, ac[s][gi], false, false);
        }
    }
#pragma unroll
    for (int s = 0; s < 2; s++)
#pragma unroll
      for (int v = 0; v < 8; v++) {
        float ig = 1.f / (1.f + expf(-ac[s][0][v]));
        float fg = 1.f / (1.f + expf(-ac[s][1][v]));
        float gg = tanhf(ac[s][2][v]);
        float og = 1.f / (1.f + expf(-ac[s][3][v]));
        float cN = fg * cst[s][v] + ig * gg;
        cst[s][v] = cN;
        float hN = og * tanhf(cN);
        int brow = v + mlo;                        // local batch row 0..15
        int n = w * 32 + s * 16 + nl;
        wB[brow * H_ + n] = (_Float16)hN;
        hD[((size_t)tt * B_ + bg * 16 + brow) * H_ + n] = hN;
      }
    __syncthreads();
  }
}

// ---------------- emissions: em[t,b,k] = [hf;hb] . w_out[k] + b_out[k]
__global__ void emissions_k(const float* __restrict__ hOut, const float* __restrict__ wOut,
                            const float* __restrict__ bOut, float* __restrict__ em) {
  __shared__ float hrow[512];
  int r = blockIdx.x, tid = threadIdx.x;
  for (int q = tid; q < 512; q += 64)
    hrow[q] = (q < 256) ? hOut[(size_t)r * H_ + q]
                        : hOut[(size_t)NROW * H_ + (size_t)r * H_ + (q - 256)];
  __syncthreads();
  if (tid < K_) {
    float s = bOut[tid];
    for (int h = 0; h < 512; ++h) s += wOut[tid * 512 + h] * hrow[h];
    em[(size_t)r * 32 + tid] = s;
  }
}

// ---------------- CRF: one wave per batch, lane-per-tag (K=20 <= 32)
__global__ void crf_nll(const float* __restrict__ em, const int* __restrict__ y,
                        const unsigned char* __restrict__ mask,
                        const float* __restrict__ startT, const float* __restrict__ endT,
                        const float* __restrict__ trans, float* __restrict__ perB) {
  __shared__ float trS[400];
  int tid = threadIdx.x;
  for (int i = tid; i < 400; i += 256) trS[i] = trans[i];
  __syncthreads();
  int lane = tid & 31, w = tid >> 5;
  int b = blockIdx.x * 8 + w;
  int k = (lane < K_) ? lane : 0;
  bool active = lane < K_;
  float alpha = active ? (startT[k] + em[(size_t)b * 32 + k]) : -1e30f;
  float score = 0.f;
  if (lane == 0) {
    int y0 = y[b * T_];
    score = startT[y0] + em[(size_t)b * 32 + y0];
  }
  for (int t = 1; t < T_; ++t) {
    float mk = mask[b * T_ + t] ? 1.f : 0.f;
    if (lane == 0) {
      int yp = y[b * T_ + t - 1], yc = y[b * T_ + t];
      score += mk * (trS[yp * 20 + yc] + em[((size_t)t * B_ + b) * 32 + yc]);
    }
    float m = -1e30f, ssum = 0.f;
    for (int j = 0; j < K_; ++j) {               // online logsumexp over prev tags
      float aj = __shfl(alpha, j, 32);
      float v = aj + trS[j * 20 + k];
      if (v > m) { ssum = ssum * expf(m - v) + 1.f; m = v; }
      else ssum += expf(v - m);
    }
    float na = m + logf(ssum) + em[((size_t)t * B_ + b) * 32 + k];
    if (active && mk > 0.f) alpha = na;
  }
  float ae = active ? (alpha + endT[k]) : -1e30f;
  float mx = ae;
  for (int off = 16; off > 0; off >>= 1) mx = fmaxf(mx, __shfl_xor(mx, off, 32));
  float ex = expf(ae - mx);
  for (int off = 16; off > 0; off >>= 1) ex += __shfl_xor(ex, off, 32);
  float logZ = mx + logf(ex);
  if (lane == 0) {
    int cnt = 0;
    for (int t = 0; t < T_; ++t) cnt += mask[b * T_ + t] ? 1 : 0;
    int lastTag = y[b * T_ + (cnt - 1)];
    score += endT[lastTag];
    perB[b] = logZ - score;
  }
}

__global__ void reduce_sum(const float* __restrict__ perB, float* __restrict__ out) {
  __shared__ float s[128];
  int tid = threadIdx.x;
  s[tid] = perB[tid];
  __syncthreads();
  for (int off = 64; off; off >>= 1) {
    if (tid < off) s[tid] += s[tid + off];
    __syncthreads();
  }
  if (tid == 0) out[0] = s[0];
}

extern "C" void kernel_launch(void* const* d_in, const int* in_sizes, int n_in,
                              void* d_out, int out_size, void* d_ws, size_t ws_size,
                              hipStream_t stream) {
  (void)in_sizes; (void)n_in; (void)out_size; (void)ws_size;
  const int* x   = (const int*)d_in[0];
  const int* y   = (const int*)d_in[1];
  const int* ptg = (const int*)d_in[2];
  const int* pyt = (const int*)d_in[3];
  const unsigned char* mask = (const unsigned char*)d_in[4];
  const float* ce   = (const float*)d_in[5];
  const float* te   = (const float*)d_in[6];
  const float* pe   = (const float*)d_in[7];
  const float* wihf = (const float*)d_in[8];
  const float* whhf = (const float*)d_in[9];
  const float* bihf = (const float*)d_in[10];
  const float* bhhf = (const float*)d_in[11];
  const float* wihb = (const float*)d_in[12];
  const float* whhb = (const float*)d_in[13];
  const float* bihb = (const float*)d_in[14];
  const float* bhhb = (const float*)d_in[15];
  const float* wOut = (const float*)d_in[16];
  const float* bOut = (const float*)d_in[17];
  const float* st   = (const float*)d_in[18];
  const float* en   = (const float*)d_in[19];
  const float* tr   = (const float*)d_in[20];

  char* ws = (char*)d_ws;
  _Float16* emb  = (_Float16*)(ws + OFF_EMB);
  _Float16* wihP = (_Float16*)(ws + OFF_WIH);
  _Float16* whhP = (_Float16*)(ws + OFF_WHH);
  float* bias = (float*)(ws + OFF_BIAS);
  float* pre  = (float*)(ws + OFF_PRE);
  float* hBuf = (float*)(ws + OFF_H);
  float* em   = (float*)(ws + OFF_EM);
  float* perB = (float*)(ws + OFF_NLL);

  embed_gather<<<(NROW * DPAD) / 256, 256, 0, stream>>>(x, ptg, pyt, ce, te, pe, emb);
  pack_wih<<<4096, 256, 0, stream>>>(wihf, wihb, wihP);
  pack_whh<<<2048, 256, 0, stream>>>(whhf, whhb, whhP);
  bias_comb<<<8, 256, 0, stream>>>(bihf, bhhf, bihb, bhhb, bias);

  int gemmLds = 2 * 128 * 512 * (int)sizeof(_Float16);   // 256 KB (A+B tiles)
  (void)hipFuncSetAttribute((const void*)gemm_inproj,
                            hipFuncAttributeMaxDynamicSharedMemorySize, gemmLds);
  gemm_inproj<<<dim3(16, 512), 256, gemmLds, stream>>>(emb, wihP, bias, pre);

  lstm_recurrent<<<16, 256, 0, stream>>>(whhP, pre, hBuf);

  emissions_k<<<NROW, 64, 0, stream>>>(hBuf, wOut, bOut, em);
  crf_nll<<<16, 256, 0, stream>>>(em, y, mask, st, en, tr, perB);
  reduce_sum<<<1, 128, 0, stream>>>(perB, (float*)d_out);
}
